// MultiHeadedAttention_7395933683922
// MI455X (gfx1250) — matrix operand order repaired and checked by reading, not compiled, not run on hardware
//
#include <hip/hip_runtime.h>
#include <hip/hip_bf16.h>

// ---------------------------------------------------------------------------
// MI455X (gfx1250, wave32) multi-head attention:
//   convert -> 3x WMMA GEMM (Q,K,V) -> flash attention (WMMA) -> WMMA GEMM (O)
// All matrix math on v_wmma_f32_16x16x32_f16 with fp32 accumulation.
// GEMM uses 32x64 register blocking per wave (8 WMMAs per K-step, 6 frag loads)
// to lift arithmetic intensity vs the L2 path (~21 FLOP/B).
// ---------------------------------------------------------------------------

typedef __attribute__((ext_vector_type(16))) _Float16 v16h;
typedef __attribute__((ext_vector_type(8)))  _Float16 v8h;
typedef __attribute__((ext_vector_type(4)))  _Float16 v4h;
typedef __attribute__((ext_vector_type(8)))  float    v8f;
typedef __attribute__((ext_vector_type(4)))  float    v4f;

union Frag16 { v16h f; v8h h[2]; };

// Two contiguous 16B runs -> one 16-half WMMA operand (matches CDNA5 16-bit
// A/B fragment layout: elements 0..7 and 8..15 are contiguous-K runs).
__device__ __forceinline__ v16h load_frag(const _Float16* p0, const _Float16* p1) {
    Frag16 fr;
    fr.h[0] = *(const v8h*)p0;
    fr.h[1] = *(const v8h*)p1;
    return fr.f;
}

__device__ __forceinline__ v8f wmma_f16(v16h a, v16h b, v8f c) {
    return __builtin_amdgcn_wmma_f32_16x16x32_f16(
        /*neg_a=*/false, a, /*neg_b=*/false, b,
        /*c_mod=*/(short)0, c, /*reuse_a=*/false, /*reuse_b=*/false);
}

// ---------------------------------------------------------------------------
// fp32 -> f16 flat convert (vectorized x4)
// ---------------------------------------------------------------------------
__global__ __launch_bounds__(256) void cvt_f32_to_f16(
    const float* __restrict__ in, _Float16* __restrict__ out, int n) {
    int i = (blockIdx.x * blockDim.x + threadIdx.x) * 4;
    if (i + 3 < n) {
        v4f v = *(const v4f*)(in + i);
        v4h h;
#pragma unroll
        for (int j = 0; j < 4; ++j) h[j] = (_Float16)v[j];
        *(v4h*)(out + i) = h;
    }
}

// ---------------------------------------------------------------------------
// fp32 W[K][N] -> f16 Wt[N][K] (transposed so GEMM B-fragments are contiguous)
// ---------------------------------------------------------------------------
__global__ __launch_bounds__(256) void cvt_transpose_f16(
    const float* __restrict__ W, _Float16* __restrict__ Wt, int K, int N) {
    int k = blockIdx.x * 16 + threadIdx.x;
    int n = blockIdx.y * 16 + threadIdx.y;
    if (k < K && n < N)
        Wt[(size_t)n * K + k] = (_Float16)W[(size_t)k * N + n];
}

// ---------------------------------------------------------------------------
// WMMA GEMM: C[M][N] = A[M][K](f16) * Bt[N][K](f16)^T + bias.
// Each wave owns a 32x64 macro-tile: 2 (M) x 4 (N) WMMA tiles, 8 fp32
// accumulator fragments held in registers across the whole K loop.
// mode 0: f16 out, (B,H,T,dk) layout, result scaled (Q gets dk^-0.5 folded in)
// mode 1: f16 out, (B,H,dk,T) layout (transposed V for contiguous P*V frags)
// mode 2: fp32 out, row-major [M][N] (final projection)
// ---------------------------------------------------------------------------
__global__ __launch_bounds__(256) void gemm16_wmma(
    const _Float16* __restrict__ A, const _Float16* __restrict__ Bt,
    const float* __restrict__ bias, _Float16* __restrict__ out16,
    float* __restrict__ out32, int M, int N, int K,
    int mode, float scale, int T, int H, int DKD) {
    const int lane = threadIdx.x & 31;
    const int wave = threadIdx.x >> 5;
    const int hi   = lane >> 4;       // which 16-lane half
    const int l15  = lane & 15;
    const int tilesN = N >> 6;                    // 64-wide wave tiles
    const int tid = blockIdx.x * 8 + wave;        // one 32x64 tile per wave
    const int tm = (tid / tilesN) * 32;
    const int tn = (tid % tilesN) * 64;

    const int akb = hi * 16;   // A: K-subblock per half
    const int bkb = hi * 16;   // B: K-subblock per half

    const _Float16* arow0 = A  + (size_t)(tm + l15)      * K;
    const _Float16* arow1 = A  + (size_t)(tm + 16 + l15) * K;
    const _Float16* brow0 = Bt + (size_t)(tn +  0 + l15) * K;
    const _Float16* brow1 = Bt + (size_t)(tn + 16 + l15) * K;
    const _Float16* brow2 = Bt + (size_t)(tn + 32 + l15) * K;
    const _Float16* brow3 = Bt + (size_t)(tn + 48 + l15) * K;

    v8f acc[2][4];
#pragma unroll
    for (int s = 0; s < 2; ++s)
#pragma unroll
        for (int j = 0; j < 4; ++j) acc[s][j] = (v8f){};

    for (int k0 = 0; k0 < K; k0 += 32) {
        const v16h a0 = load_frag(arow0 + k0 + akb, arow0 + k0 + akb + 8);
        const v16h a1 = load_frag(arow1 + k0 + akb, arow1 + k0 + akb + 8);
        const v16h b0 = load_frag(brow0 + k0 + bkb, brow0 + k0 + bkb + 8);
        const v16h b1 = load_frag(brow1 + k0 + bkb, brow1 + k0 + bkb + 8);
        const v16h b2 = load_frag(brow2 + k0 + bkb, brow2 + k0 + bkb + 8);
        const v16h b3 = load_frag(brow3 + k0 + bkb, brow3 + k0 + bkb + 8);
        __builtin_prefetch(arow0 + k0 + 64, 0, 1);   // global_prefetch_b8
        __builtin_prefetch(arow1 + k0 + 64, 0, 1);
        __builtin_prefetch(brow0 + k0 + 64, 0, 1);
        __builtin_prefetch(brow2 + k0 + 64, 0, 1);
        acc[0][0] = wmma_f16(a0, b0, acc[0][0]);
        acc[0][1] = wmma_f16(a0, b1, acc[0][1]);
        acc[0][2] = wmma_f16(a0, b2, acc[0][2]);
        acc[0][3] = wmma_f16(a0, b3, acc[0][3]);
        acc[1][0] = wmma_f16(a1, b0, acc[1][0]);
        acc[1][1] = wmma_f16(a1, b1, acc[1][1]);
        acc[1][2] = wmma_f16(a1, b2, acc[1][2]);
        acc[1][3] = wmma_f16(a1, b3, acc[1][3]);
    }

#pragma unroll
    for (int j = 0; j < 4; ++j) {
        const int col = tn + j * 16 + l15;
        const float bb = bias[col];
        const int hh = col / DKD, dd = col % DKD;
#pragma unroll
        for (int s = 0; s < 2; ++s) {
#pragma unroll
            for (int i = 0; i < 8; ++i) {
                const int mg = tm + s * 16 + i + 8 * hi;   // global row (C layout)
                const float val = (acc[s][j][i] + bb) * scale;
                if (mode == 2) {
                    out32[(size_t)mg * N + col] = val;
                } else {
                    const int bidx = mg / T, tt = mg % T;
                    size_t idx;
                    if (mode == 0)
                        idx = ((size_t)(bidx * H + hh) * T + tt) * DKD + dd;
                    else
                        idx = ((size_t)(bidx * H + hh) * DKD + dd) * (size_t)T + tt;
                    out16[idx] = (_Float16)val;
                }
            }
        }
    }
}

// ---------------------------------------------------------------------------
// Flash attention. Grid: (T/64, B*H). Block: 128 threads = 4 waves.
// Each wave owns 16 query rows; streams 32-key tiles with online softmax.
// Q,K in (b,h,T,dk) f16; V transposed (b,h,dk,T) f16; out (b,T,H*dk) f16.
// ---------------------------------------------------------------------------
#define FT  2048
#define FDK 64
#define FH  16

__global__ __launch_bounds__(128) void flash_attn_wmma(
    const _Float16* __restrict__ Q, const _Float16* __restrict__ Kh,
    const _Float16* __restrict__ Vt, const unsigned char* __restrict__ pm,
    _Float16* __restrict__ O) {
    // Per-wave 16x32 P staging tile; row stride 48 halves (96B, 16B aligned)
    __shared__ _Float16 ldsP[4 * 16 * 48];

    const int lane = threadIdx.x & 31;
    const int wave = threadIdx.x >> 5;
    const int hi   = lane >> 4;
    const int l15  = lane & 15;
    const int bh   = blockIdx.y;           // b*H + h
    const int b    = bh / FH;
    const int h    = bh % FH;
    const int q0   = blockIdx.x * 64 + wave * 16;
    _Float16* pw = ldsP + wave * (16 * 48);

    const _Float16* qbase = Q  + (size_t)bh * FT * FDK;
    const _Float16* kbase = Kh + (size_t)bh * FT * FDK;
    const _Float16* vbase = Vt + (size_t)bh * FDK * FT;
    const unsigned char* pmb = pm + (size_t)b * FT * FT;

    const int akb = hi * 16;
    const int bkb = hi * 16;

    // Preload Q A-fragments for dk chunks [0,32) and [32,64)
    const _Float16* qrow = qbase + (size_t)(q0 + l15) * FDK;
    const v16h aq0 = load_frag(qrow + akb,      qrow + akb + 8);
    const v16h aq1 = load_frag(qrow + 32 + akb, qrow + 32 + akb + 8);

    v8f o0 = {}, o1 = {}, o2 = {}, o3 = {};
    v8f mrun, lrun;
#pragma unroll
    for (int i = 0; i < 8; ++i) { mrun[i] = -1e30f; lrun[i] = 0.f; }

    const int ntiles = 2 * blockIdx.x + 2;   // block-uniform (barriers legal)
    for (int t = 0; t < ntiles; ++t) {
        const int kt0 = t * 32;

        // ---- S = Q * K^T for 32 keys (two 16-key N halves) ----
        const _Float16* krow0 = kbase + (size_t)(kt0 + l15) * FDK;
        const _Float16* krow1 = krow0 + 16 * FDK;
        const v16h bk00 = load_frag(krow0 + bkb,      krow0 + bkb + 8);
        const v16h bk01 = load_frag(krow0 + 32 + bkb, krow0 + 32 + bkb + 8);
        const v16h bk10 = load_frag(krow1 + bkb,      krow1 + bkb + 8);
        const v16h bk11 = load_frag(krow1 + 32 + bkb, krow1 + 32 + bkb + 8);
        v8f zero = {};
        v8f s0 = wmma_f16(aq1, bk01, wmma_f16(aq0, bk00, zero));
        v8f s1 = wmma_f16(aq1, bk11, wmma_f16(aq0, bk10, zero));

        // ---- masking: padding mask + causal (row 0 = CLS attends all) ----
        const int col0 = kt0 + l15;
        const int col1 = col0 + 16;
#pragma unroll
        for (int i = 0; i < 8; ++i) {
            const int qr = q0 + i + 8 * hi;
            const unsigned char* pr = pmb + (size_t)qr * FT;
            const bool a0 = pr[col0] && (qr == 0 || col0 <= qr);
            const bool a1 = pr[col1] && (qr == 0 || col1 <= qr);
            s0[i] = a0 ? s0[i] : -1e30f;
            s1[i] = a1 ? s1[i] : -1e30f;
        }

        // ---- online softmax: row max across the 16-lane half ----
        v8f tmax;
#pragma unroll
        for (int i = 0; i < 8; ++i) tmax[i] = fmaxf(s0[i], s1[i]);
#pragma unroll
        for (int msk = 1; msk <= 8; msk <<= 1) {
#pragma unroll
            for (int i = 0; i < 8; ++i)
                tmax[i] = fmaxf(tmax[i], __shfl_xor(tmax[i], msk, 32));
        }

        v8f corr, rsum;
#pragma unroll
        for (int i = 0; i < 8; ++i) {
            const float nm = fmaxf(mrun[i], tmax[i]);
            corr[i] = __expf(mrun[i] - nm);
            s0[i]   = __expf(s0[i] - nm);
            s1[i]   = __expf(s1[i] - nm);
            rsum[i] = s0[i] + s1[i];
            mrun[i] = nm;
        }
#pragma unroll
        for (int msk = 1; msk <= 8; msk <<= 1) {
#pragma unroll
            for (int i = 0; i < 8; ++i)
                rsum[i] += __shfl_xor(rsum[i], msk, 32);
        }
#pragma unroll
        for (int i = 0; i < 8; ++i) {
            lrun[i] = lrun[i] * corr[i] + rsum[i];
            o0[i] *= corr[i]; o1[i] *= corr[i];
            o2[i] *= corr[i]; o3[i] *= corr[i];
        }

        // ---- stage P (16x32, C layout) through LDS, reload as A fragment ----
#pragma unroll
        for (int i = 0; i < 8; ++i) {
            const int r = i + 8 * hi;
            pw[r * 48 + l15]      = (_Float16)s0[i];
            pw[r * 48 + 16 + l15] = (_Float16)s1[i];
        }
        __syncthreads();
        const _Float16* prow = pw + l15 * 48;
        const v16h ap = load_frag(prow + akb, prow + akb + 8);

        // ---- O += P * V  (V transposed: contiguous key runs per dim) ----
        const _Float16* v0 = vbase + (size_t)(0 * 16 + l15) * FT + kt0;
        const _Float16* v1 = vbase + (size_t)(1 * 16 + l15) * FT + kt0;
        const _Float16* v2 = vbase + (size_t)(2 * 16 + l15) * FT + kt0;
        const _Float16* v3 = vbase + (size_t)(3 * 16 + l15) * FT + kt0;
        o0 = wmma_f16(ap, load_frag(v0 + bkb, v0 + bkb + 8), o0);
        o1 = wmma_f16(ap, load_frag(v1 + bkb, v1 + bkb + 8), o1);
        o2 = wmma_f16(ap, load_frag(v2 + bkb, v2 + bkb + 8), o2);
        o3 = wmma_f16(ap, load_frag(v3 + bkb, v3 + bkb + 8), o3);
        __syncthreads();
    }

    // ---- normalize by row sum, store f16 in (B, T, H*dk) for final GEMM ----
#pragma unroll
    for (int i = 0; i < 8; ++i) {
        const float inv = 1.0f / lrun[i];
        const int tq = q0 + i + 8 * hi;
        const size_t ro = ((size_t)b * FT + tq) * (FH * FDK) + h * FDK;
        O[ro + 0 * 16 + l15] = (_Float16)(o0[i] * inv);
        O[ro + 1 * 16 + l15] = (_Float16)(o1[i] * inv);
        O[ro + 2 * 16 + l15] = (_Float16)(o2[i] * inv);
        O[ro + 3 * 16 + l15] = (_Float16)(o3[i] * inv);
    }
}

// ---------------------------------------------------------------------------
extern "C" void kernel_launch(void* const* d_in, const int* in_sizes, int n_in,
                              void* d_out, int out_size, void* d_ws, size_t ws_size,
                              hipStream_t stream) {
    (void)in_sizes; (void)n_in; (void)out_size; (void)ws_size;
    const float* x            = (const float*)d_in[0];
    const unsigned char* pmsk = (const unsigned char*)d_in[1];
    const float* Wq = (const float*)d_in[2];
    const float* bq = (const float*)d_in[3];
    const float* Wk = (const float*)d_in[4];
    const float* bk = (const float*)d_in[5];
    const float* Wv = (const float*)d_in[6];
    const float* bv = (const float*)d_in[7];
    const float* Wo = (const float*)d_in[8];
    const float* bo = (const float*)d_in[9];
    float* out = (float*)d_out;

    const int Bb = 2, T = 2048, D = 1024, H = 16, DK = 64;
    const int M = Bb * T;                       // 4096

    // workspace carve-up (all f16)
    char* ws = (char*)d_ws;
    _Float16* x16  = (_Float16*)ws; ws += (size_t)M * D * 2;
    _Float16* wqt  = (_Float16*)ws; ws += (size_t)D * D * 2;
    _Float16* wkt  = (_Float16*)ws; ws += (size_t)D * D * 2;
    _Float16* wvt  = (_Float16*)ws; ws += (size_t)D * D * 2;
    _Float16* wot  = (_Float16*)ws; ws += (size_t)D * D * 2;
    _Float16* qh   = (_Float16*)ws; ws += (size_t)M * D * 2;   // (B,H,T,dk)
    _Float16* kh   = (_Float16*)ws; ws += (size_t)M * D * 2;   // (B,H,T,dk)
    _Float16* vt   = (_Float16*)ws; ws += (size_t)M * D * 2;   // (B,H,dk,T)
    _Float16* attn = (_Float16*)ws; ws += (size_t)M * D * 2;   // (B,T,H*dk)

    // 1) conversions
    cvt_f32_to_f16<<<(M * D) / (4 * 256), 256, 0, stream>>>(x, x16, M * D);
    dim3 tg(D / 16, D / 16), tb(16, 16);
    cvt_transpose_f16<<<tg, tb, 0, stream>>>(Wq, wqt, D, D);
    cvt_transpose_f16<<<tg, tb, 0, stream>>>(Wk, wkt, D, D);
    cvt_transpose_f16<<<tg, tb, 0, stream>>>(Wv, wvt, D, D);
    cvt_transpose_f16<<<tg, tb, 0, stream>>>(Wo, wot, D, D);

    // 2) QKV projections (scale dk^-0.5 folded into Q)
    // one 32x64 macro-tile per wave, 8 waves per block
    const int gblocks = (M / 32) * (D / 64) / 8;
    gemm16_wmma<<<gblocks, 256, 0, stream>>>(x16, wqt, bq, qh, nullptr,
                                             M, D, D, 0, 0.125f, T, H, DK);
    gemm16_wmma<<<gblocks, 256, 0, stream>>>(x16, wkt, bk, kh, nullptr,
                                             M, D, D, 0, 1.0f, T, H, DK);
    gemm16_wmma<<<gblocks, 256, 0, stream>>>(x16, wvt, bv, vt, nullptr,
                                             M, D, D, 1, 1.0f, T, H, DK);

    // 3) flash attention
    flash_attn_wmma<<<dim3(T / 64, Bb * H), 128, 0, stream>>>(qh, kh, vt, pmsk, attn);

    // 4) output projection -> fp32 d_out
    gemm16_wmma<<<gblocks, 256, 0, stream>>>(attn, wot, bo, nullptr, out,
                                             M, D, D, 2, 1.0f, T, H, DK);
}
